// Net_60550448939587
// MI455X (gfx1250) — compile-verified, matrix operand order checked by reading, not run on hardware
//
#include <hip/hip_runtime.h>
#include <hip/hip_bf16.h>
#include <math.h>

#define NNODE 100000
#define FIN   512
#define HD    64      // H*D layer-1 output width
#define NH    8       // heads layer 1
#define DH    8       // per-head dim
#define CC    40      // classes
#define NEGSL 0.2f

typedef __attribute__((ext_vector_type(16))) __bf16 v16bf;
typedef __attribute__((ext_vector_type(8)))  float  v8f;

static __host__ __device__ inline int cdiv(int a, int b) { return (a + b - 1) / b; }

// ---------------------------------------------------------------- utilities
__global__ void fill_f32(float* __restrict__ p, long n, float v) {
    long i = (long)blockIdx.x * blockDim.x + threadIdx.x;
    if (i < n) p[i] = v;
}

__device__ inline void pack8(v16bf& a, int base, const float4& lo, const float4& hi) {
    a[base + 0] = (__bf16)lo.x; a[base + 1] = (__bf16)lo.y;
    a[base + 2] = (__bf16)lo.z; a[base + 3] = (__bf16)lo.w;
    a[base + 4] = (__bf16)hi.x; a[base + 5] = (__bf16)hi.y;
    a[base + 6] = (__bf16)hi.z; a[base + 7] = (__bf16)hi.w;
}

// ---------------------------------------------------------------- GEMM 1
// h1pre[N,64] = x[N,512] @ W1[512,64].
// One wave -> full 16x64 row block: A fragment loaded ONCE per K-step
// (b128 vector loads), reused across 4 N-tiles (4 consecutive WMMAs).
// block = 32 (one wave, EXEC all ones as WMMA requires).
__global__ __launch_bounds__(32) void gemm1_wmma(const float* __restrict__ x,
                                                 const float* __restrict__ W,
                                                 float* __restrict__ out) {
    const int lane = threadIdx.x & 31;
    const int half = lane >> 4;                 // K-half selector
    const int l16  = lane & 15;
    const int mrow = blockIdx.x * 16 + l16;     // A row this lane feeds
    const float4* xrow = (const float4*)(x + (size_t)mrow * FIN);

    v8f c0 = {}, c1 = {}, c2 = {}, c3 = {};
    for (int kb = 0; kb < FIN; kb += 32) {
        // ---- A fragment: two contiguous 8-float runs, 32B aligned
        const int k0 = kb + (half << 3);        // elements j=0..7
        const int k1 = kb + 16 + (half << 3);   // elements j=8..15
        const float4 a0 = xrow[(k0 >> 2)];
        const float4 a1 = xrow[(k0 >> 2) + 1];
        const float4 a2 = xrow[(k1 >> 2)];
        const float4 a3 = xrow[(k1 >> 2) + 1];
        v16bf a;
        pack8(a, 0, a0, a1);
        pack8(a, 8, a2, a3);
        // ---- 4 B fragments (W1 is L2-resident), 4 WMMAs sharing A
        v8f* cs[4] = { &c0, &c1, &c2, &c3 };
#pragma unroll
        for (int t = 0; t < 4; ++t) {
            v16bf b;
#pragma unroll
            for (int j = 0; j < 16; ++j) {
                const int k = kb + ((j >> 3) << 4) + (half << 3) + (j & 7);
                b[j] = (__bf16)W[(size_t)k * HD + t * 16 + l16];
            }
            *cs[t] = __builtin_amdgcn_wmma_f32_16x16x32_bf16(
                false, a, false, b, (short)0, *cs[t], false, false);
        }
    }
    // ---- store 16x64 block; adjacent lanes -> contiguous columns
    const v8f cc[4] = { c0, c1, c2, c3 };
#pragma unroll
    for (int t = 0; t < 4; ++t)
#pragma unroll
        for (int r = 0; r < 8; ++r) {
            const int mm = blockIdx.x * 16 + r + half * 8;
            out[(size_t)mm * HD + t * 16 + l16] = cc[t][r];
        }
}

// ---------------------------------------------------------------- GEMM 2
// h2pre[N,40] = h1post[N,64] @ W2[64,40]; 3 N-tiles cover 48 >= 40,
// A fragment shared across tiles, K=64 -> 2 WMMA steps per tile.
__global__ __launch_bounds__(32) void gemm2_wmma(const float* __restrict__ x,
                                                 const float* __restrict__ W,
                                                 float* __restrict__ out) {
    const int lane = threadIdx.x & 31;
    const int half = lane >> 4;
    const int l16  = lane & 15;
    const int mrow = blockIdx.x * 16 + l16;
    const float4* xrow = (const float4*)(x + (size_t)mrow * HD);

    v8f c0 = {}, c1 = {}, c2 = {};
#pragma unroll
    for (int kb = 0; kb < HD; kb += 32) {
        const int k0 = kb + (half << 3);
        const int k1 = kb + 16 + (half << 3);
        const float4 a0 = xrow[(k0 >> 2)];
        const float4 a1 = xrow[(k0 >> 2) + 1];
        const float4 a2 = xrow[(k1 >> 2)];
        const float4 a3 = xrow[(k1 >> 2) + 1];
        v16bf a;
        pack8(a, 0, a0, a1);
        pack8(a, 8, a2, a3);
        v8f* cs[3] = { &c0, &c1, &c2 };
#pragma unroll
        for (int t = 0; t < 3; ++t) {
            const int ncol = t * 16 + l16;
            v16bf b;
#pragma unroll
            for (int j = 0; j < 16; ++j) {
                const int k = kb + ((j >> 3) << 4) + (half << 3) + (j & 7);
                b[j] = (ncol < CC) ? (__bf16)W[(size_t)k * CC + ncol] : (__bf16)0.0f;
            }
            *cs[t] = __builtin_amdgcn_wmma_f32_16x16x32_bf16(
                false, a, false, b, (short)0, *cs[t], false, false);
        }
    }
    const v8f cc[3] = { c0, c1, c2 };
#pragma unroll
    for (int t = 0; t < 3; ++t)
#pragma unroll
        for (int r = 0; r < 8; ++r) {
            const int mm = blockIdx.x * 16 + r + half * 8;
            const int nn = t * 16 + l16;
            if (nn < CC) out[(size_t)mm * CC + nn] = cc[t][r];
        }
}

// ---------------------------------------------------------------- attention logit dots
__global__ void alpha1_kernel(const float* __restrict__ h,
                              const float* __restrict__ asrc,
                              const float* __restrict__ adst,
                              float* __restrict__ as_out,
                              float* __restrict__ ad_out) {
    int idx = blockIdx.x * blockDim.x + threadIdx.x;
    if (idx >= NNODE * NH) return;
    const int n = idx >> 3, hh = idx & 7;
    const float4* hp = (const float4*)(h + (size_t)n * HD + hh * DH);
    const float4 v0 = hp[0], v1 = hp[1];
    float s = 0.f, d = 0.f;
    const float* va = &v0.x;  // 8 contiguous values v0,v1
    float vv[8] = { v0.x, v0.y, v0.z, v0.w, v1.x, v1.y, v1.z, v1.w };
    (void)va;
#pragma unroll
    for (int k = 0; k < DH; ++k) {
        s += vv[k] * asrc[hh * DH + k];
        d += vv[k] * adst[hh * DH + k];
    }
    as_out[idx] = s;
    ad_out[idx] = d;
}

__global__ void alpha2_kernel(const float* __restrict__ h,
                              const float* __restrict__ asrc,
                              const float* __restrict__ adst,
                              float* __restrict__ as_out,
                              float* __restrict__ ad_out) {
    int n = blockIdx.x * blockDim.x + threadIdx.x;
    if (n >= NNODE) return;
    float s = 0.f, d = 0.f;
#pragma unroll
    for (int c = 0; c < CC; ++c) {
        const float v = h[(size_t)n * CC + c];
        s += v * asrc[c];
        d += v * adst[c];
    }
    as_out[n] = s;
    ad_out[n] = d;
}

__device__ inline float leaky(float v) { return v >= 0.f ? v : NEGSL * v; }

__device__ inline void edge_endpoints(const int* __restrict__ ei, int e, int E,
                                      int& src, int& dst) {
    if (e < E) { src = ei[e]; dst = ei[E + e]; }
    else       { src = dst = e - E; }          // self loops
}

// ---------------------------------------------------------------- layer-1 edge passes
__global__ void edge1_max(const int* __restrict__ ei, int E, int Etot,
                          const float* __restrict__ as1,
                          const float* __restrict__ ad1,
                          float* __restrict__ emax) {
    int idx = blockIdx.x * blockDim.x + threadIdx.x;
    if (idx >= Etot * NH) return;
    const int e = idx >> 3, hh = idx & 7;
    int src, dst; edge_endpoints(ei, e, E, src, dst);
    const float logit = leaky(as1[src * NH + hh] + ad1[dst * NH + hh]);
    atomicMax(&emax[dst * NH + hh], logit);
}

__global__ void edge1_agg(const int* __restrict__ ei, int E, int Etot,
                          const float* __restrict__ as1,
                          const float* __restrict__ ad1,
                          const float* __restrict__ emax,
                          const float* __restrict__ h1,
                          float* __restrict__ denom,
                          float* __restrict__ acc) {
    int idx = blockIdx.x * blockDim.x + threadIdx.x;
    if (idx >= Etot * NH) return;
    const int e = idx >> 3, hh = idx & 7;
    int src, dst; edge_endpoints(ei, e, E, src, dst);
    const float logit = leaky(as1[src * NH + hh] + ad1[dst * NH + hh]);
    const float ee = expf(logit - emax[dst * NH + hh]);
    atomicAdd(&denom[dst * NH + hh], ee);
    // vectorized gather of the 8-float head slice (L2-resident)
    const float4* hp = (const float4*)(h1 + (size_t)src * HD + hh * DH);
    const float4 v0 = hp[0], v1 = hp[1];
    const float vv[8] = { v0.x, v0.y, v0.z, v0.w, v1.x, v1.y, v1.z, v1.w };
    float* ap = acc + (size_t)dst * HD + hh * DH;
#pragma unroll
    for (int d = 0; d < DH; ++d)
        atomicAdd(ap + d, ee * vv[d]);
}

// normalize, +bias, ELU
__global__ void finalize1(const float* __restrict__ acc,
                          const float* __restrict__ denom,
                          const float* __restrict__ b1,
                          float* __restrict__ h1post) {
    int idx = blockIdx.x * blockDim.x + threadIdx.x;
    if (idx >= NNODE * HD) return;
    const int n = idx >> 6, hd = idx & 63, hh = hd >> 3;
    float v = acc[idx] / (denom[n * NH + hh] + 1e-16f) + b1[hd];
    h1post[idx] = v > 0.f ? v : expm1f(v);
}

// ---------------------------------------------------------------- layer-2 edge passes
__global__ void edge2_max(const int* __restrict__ ei, int E, int Etot,
                          const float* __restrict__ as2,
                          const float* __restrict__ ad2,
                          float* __restrict__ emax) {
    int e = blockIdx.x * blockDim.x + threadIdx.x;
    if (e >= Etot) return;
    int src, dst; edge_endpoints(ei, e, E, src, dst);
    atomicMax(&emax[dst], leaky(as2[src] + ad2[dst]));
}

__global__ void edge2_agg(const int* __restrict__ ei, int E, int Etot,
                          const float* __restrict__ as2,
                          const float* __restrict__ ad2,
                          const float* __restrict__ emax,
                          const float* __restrict__ h2,
                          float* __restrict__ denom,
                          float* __restrict__ acc) {
    int e = blockIdx.x * blockDim.x + threadIdx.x;
    if (e >= Etot) return;
    int src, dst; edge_endpoints(ei, e, E, src, dst);
    const float ee = expf(leaky(as2[src] + ad2[dst]) - emax[dst]);
    atomicAdd(&denom[dst], ee);
#pragma unroll
    for (int c = 0; c < CC; ++c)
        atomicAdd(&acc[(size_t)dst * CC + c], ee * h2[(size_t)src * CC + c]);
}

// normalize, +bias, log_softmax over 40 classes
__global__ void finalize2(const float* __restrict__ acc,
                          const float* __restrict__ denom,
                          const float* __restrict__ b2,
                          float* __restrict__ out) {
    int n = blockIdx.x * blockDim.x + threadIdx.x;
    if (n >= NNODE) return;
    float vals[CC];
    const float inv = 1.f / (denom[n] + 1e-16f);
    float mx = -INFINITY;
#pragma unroll
    for (int c = 0; c < CC; ++c) {
        vals[c] = acc[(size_t)n * CC + c] * inv + b2[c];
        mx = fmaxf(mx, vals[c]);
    }
    float s = 0.f;
#pragma unroll
    for (int c = 0; c < CC; ++c) s += expf(vals[c] - mx);
    const float lse = mx + logf(s);
#pragma unroll
    for (int c = 0; c < CC; ++c) out[(size_t)n * CC + c] = vals[c] - lse;
}

// ---------------------------------------------------------------- launch
extern "C" void kernel_launch(void* const* d_in, const int* in_sizes, int n_in,
                              void* d_out, int out_size, void* d_ws, size_t ws_size,
                              hipStream_t stream) {
    const float* x     = (const float*)d_in[0];
    const int*   ei    = (const int*)  d_in[1];
    const float* W1    = (const float*)d_in[2];
    const float* asrc1 = (const float*)d_in[3];
    const float* adst1 = (const float*)d_in[4];
    const float* b1    = (const float*)d_in[5];
    const float* W2    = (const float*)d_in[6];
    const float* asrc2 = (const float*)d_in[7];
    const float* adst2 = (const float*)d_in[8];
    const float* b2    = (const float*)d_in[9];
    float* out = (float*)d_out;

    const int E    = in_sizes[1] / 2;
    const int Etot = E + NNODE;

    // workspace layout (floats)
    float* ws     = (float*)d_ws;
    float* h1pre  = ws;                            // N*64
    float* as1    = h1pre  + (size_t)NNODE * HD;   // N*8
    float* ad1    = as1    + (size_t)NNODE * NH;
    float* emax1  = ad1    + (size_t)NNODE * NH;
    float* den1   = emax1  + (size_t)NNODE * NH;
    float* acc1   = den1   + (size_t)NNODE * NH;   // N*64
    float* h1post = acc1   + (size_t)NNODE * HD;   // N*64
    float* h2pre  = h1post + (size_t)NNODE * HD;   // N*40
    float* as2    = h2pre  + (size_t)NNODE * CC;   // N
    float* ad2    = as2    + (size_t)NNODE;
    float* emax2  = ad2    + (size_t)NNODE;
    float* den2   = emax2  + (size_t)NNODE;
    float* acc2   = den2   + (size_t)NNODE;        // N*40

    const int TB = 256;
    // per-call (re)initialization — graph-replay safe, no persistent state
    fill_f32<<<cdiv(NNODE * NH, TB), TB, 0, stream>>>(emax1, (long)NNODE * NH, -INFINITY);
    fill_f32<<<cdiv(NNODE * NH, TB), TB, 0, stream>>>(den1,  (long)NNODE * NH, 0.f);
    fill_f32<<<cdiv(NNODE * HD, TB), TB, 0, stream>>>(acc1,  (long)NNODE * HD, 0.f);
    fill_f32<<<cdiv(NNODE, TB),      TB, 0, stream>>>(emax2, (long)NNODE, -INFINITY);
    fill_f32<<<cdiv(NNODE, TB),      TB, 0, stream>>>(den2,  (long)NNODE, 0.f);
    fill_f32<<<cdiv(NNODE * CC, TB), TB, 0, stream>>>(acc2,  (long)NNODE * CC, 0.f);

    // ---- layer 1
    gemm1_wmma<<<NNODE / 16, 32, 0, stream>>>(x, W1, h1pre);
    alpha1_kernel<<<cdiv(NNODE * NH, TB), TB, 0, stream>>>(h1pre, asrc1, adst1, as1, ad1);
    edge1_max<<<cdiv(Etot * NH, TB), TB, 0, stream>>>(ei, E, Etot, as1, ad1, emax1);
    edge1_agg<<<cdiv(Etot * NH, TB), TB, 0, stream>>>(ei, E, Etot, as1, ad1, emax1,
                                                      h1pre, den1, acc1);
    finalize1<<<cdiv(NNODE * HD, TB), TB, 0, stream>>>(acc1, den1, b1, h1post);

    // ---- layer 2
    gemm2_wmma<<<NNODE / 16, 32, 0, stream>>>(h1post, W2, h2pre);
    alpha2_kernel<<<cdiv(NNODE, TB), TB, 0, stream>>>(h2pre, asrc2, adst2, as2, ad2);
    edge2_max<<<cdiv(Etot, TB), TB, 0, stream>>>(ei, E, Etot, as2, ad2, emax2);
    edge2_agg<<<cdiv(Etot, TB), TB, 0, stream>>>(ei, E, Etot, as2, ad2, emax2,
                                                 h2pre, den2, acc2);
    finalize2<<<cdiv(NNODE, TB), TB, 0, stream>>>(acc2, den2, b2, out);
}
